// MambaLM_73950746902578
// MI455X (gfx1250) — matrix operand order repaired and checked by reading, not compiled, not run on hardware
//
#include <hip/hip_runtime.h>
#include <hip/hip_bf16.h>
#include <cstddef>
#include <cstdint>

// ---------------------------------------------------------------------------
// MI455X (gfx1250) Mamba-LM forward.
//  - bf16 WMMA (16x16x32), f32 accum; each wave computes a 16x(16*NSUB) strip.
//  - K templated + explicit double-buffered fragments: next K-chunk's loads
//    issue under current chunk's 4 WMMAs (breaks s_wait_loadcnt 0x0 serialize).
//  - chunked (3-pass) selective scan: 16 chunks of 128 steps -> 1024 blocks.
// ---------------------------------------------------------------------------

typedef __attribute__((ext_vector_type(16))) __bf16 v16bf;
typedef __attribute__((ext_vector_type(8)))  float  v8f;

static constexpr int B_     = 64;
static constexpr int L_     = 2048;
static constexpr int M_     = B_ * L_;       // 131072 rows
static constexpr int DMODEL = 128;
static constexpr int DINNER = 256;
static constexpr int DSTATE = 16;
static constexpr int NL     = 4;
static constexpr int XPN    = 48;            // x_proj N padded 40 -> 48
static constexpr int NC     = 16;            // scan chunks
static constexpr int CH     = L_ / NC;       // 128 steps per chunk

__device__ __forceinline__ __bf16 f2bf(float f) {
    union { float f; unsigned u; } in; in.f = f;
    unsigned r = (in.u + 0x7FFFu + ((in.u >> 16) & 1u)) >> 16;
    union { unsigned short s; __bf16 b; } out; out.s = (unsigned short)r;
    return out.b;
}
__device__ __forceinline__ float bf2f(__bf16 b) {
    union { unsigned short s; __bf16 b; } in; in.b = b;
    union { unsigned u; float f; } out; out.u = ((unsigned)in.s) << 16;
    return out.f;
}

union FragB { v16bf v; uint4 q[2]; };
union FragC { v8f  v; float f[8]; };

// ---------------------------------------------------------------------------
// D[M,N] = A[M,K] * W[N,K]^T, bf16 in / f32 accum. Wave -> 16 x (16*NSUB).
// K is compile-time; fully unrolled, software-pipelined (2-stage).
// ---------------------------------------------------------------------------
template<int K, int NSUB, bool OUT_BF16, bool ADD_RES>
__global__ __launch_bounds__(256)
void gemm_wmma_bf16(const __bf16* __restrict__ A, const __bf16* __restrict__ W,
                    void* __restrict__ Cout, const float* __restrict__ Res,
                    int M, int N)
{
    const int lane   = threadIdx.x & 31;
    const int tile   = blockIdx.x * 8 + (threadIdx.x >> 5);
    const int tilesN = N / (16 * NSUB);
    const int tilesT = (M >> 4) * tilesN;
    if (tile >= tilesT) return;                 // wave-uniform exit
    const int tm = tile / tilesN;
    const int tn = tile - tm * tilesN;

    const int khalf = lane >> 4;                // K-half select (contiguous k = 16*khalf + e)
    const int l15   = lane & 15;
    const __bf16* arow = A + (size_t)(tm * 16 + l15) * K;
    const __bf16* wrow[NSUB];
#pragma unroll
    for (int j = 0; j < NSUB; ++j)
        wrow[j] = W + (size_t)((tn * NSUB + j) * 16 + l15) * K;

    FragC acc[NSUB];
#pragma unroll
    for (int j = 0; j < NSUB; ++j)
#pragma unroll
        for (int r = 0; r < 8; ++r) acc[j].f[r] = 0.0f;

    constexpr int NK = K / 32;
    FragB a[2], b[2][NSUB];

    // stage 0: k0 = 0
    // A 16x32 bf16: K = k0 + 16*khalf + e. Two b128s.
    a[0].q[0] = *(const uint4*)(arow + 16 * khalf);
    a[0].q[1] = *(const uint4*)(arow + 16 * khalf + 8);
#pragma unroll
    for (int j = 0; j < NSUB; ++j) {
        // B 32x16 bf16: n = lane&15, K = k0 + 16*khalf + e. 32B contiguous.
        b[0][j].q[0] = *(const uint4*)(wrow[j] + 16 * khalf);
        b[0][j].q[1] = *(const uint4*)(wrow[j] + 16 * khalf + 8);
    }

#pragma unroll
    for (int kk = 0; kk < NK; ++kk) {
        const int cur = kk & 1, nxt = cur ^ 1;
        if (kk + 1 < NK) {                       // issue next chunk's loads now
            const int k1 = (kk + 1) * 32;
            a[nxt].q[0] = *(const uint4*)(arow + k1 + 16 * khalf);
            a[nxt].q[1] = *(const uint4*)(arow + k1 + 16 * khalf + 8);
#pragma unroll
            for (int j = 0; j < NSUB; ++j) {
                b[nxt][j].q[0] = *(const uint4*)(wrow[j] + k1 + 16 * khalf);
                b[nxt][j].q[1] = *(const uint4*)(wrow[j] + k1 + 16 * khalf + 8);
            }
        }
#pragma unroll
        for (int j = 0; j < NSUB; ++j)
            acc[j].v = __builtin_amdgcn_wmma_f32_16x16x32_bf16(
                false, a[cur].v, false, b[cur][j].v, (short)0, acc[j].v,
                false, false);
    }

#pragma unroll
    for (int j = 0; j < NSUB; ++j) {
        const int col = (tn * NSUB + j) * 16 + l15;
#pragma unroll
        for (int r = 0; r < 8; ++r) {
            const int row = tm * 16 + r + 8 * khalf;   // lanes16-31 -> M=r+8
            const size_t idx = (size_t)row * N + col;
            float v = acc[j].f[r];
            if (ADD_RES) v += Res[idx];
            if (OUT_BF16) ((__bf16*)Cout)[idx] = f2bf(v);
            else          ((float*)Cout)[idx]  = v;
        }
    }
}

// -------- weight prep: f32->bf16 (+pad x_proj 40->48), A=-exp, convW transpose
__global__ void k_prep(const float* __restrict__ ipw, const float* __restrict__ xpw,
                       const float* __restrict__ opw, const float* __restrict__ alog,
                       const float* __restrict__ cw,
                       __bf16* __restrict__ wip, __bf16* __restrict__ wxp,
                       __bf16* __restrict__ wop, float* __restrict__ aneg,
                       float* __restrict__ cwt)
{
    int t = blockIdx.x * blockDim.x + threadIdx.x;
    const int N_IP = NL * 512 * 128;
    const int N_XP = NL * XPN * 256;
    const int N_OP = NL * 128 * 256;
    const int N_A  = NL * 256 * 16;
    const int N_CW = 3 * 128 * 128;
    if (t < N_IP) { wip[t] = f2bf(ipw[t]); return; }
    t -= N_IP;
    if (t < N_XP) {
        const int layer = t / (XPN * 256);
        const int rem   = t % (XPN * 256);
        const int row   = rem / 256, col = rem % 256;
        wxp[t] = f2bf(row < 40 ? xpw[(layer * 40 + row) * 256 + col] : 0.0f);
        return;
    }
    t -= N_XP;
    if (t < N_OP) { wop[t] = f2bf(opw[t]); return; }
    t -= N_OP;
    if (t < N_A)  { aneg[t] = -__expf(alog[t]); return; }
    t -= N_A;
    if (t < N_CW) {                 // [o][i][k] -> [k][o][i] for contiguous dot
        const int k = t / (128 * 128);
        const int rem = t % (128 * 128);
        const int o = rem / 128, i = rem % 128;
        cwt[t] = cw[(o * 128 + i) * 3 + k];
    }
}

// ---------------- embedding + (identity-stats) batchnorm ---------------------
__global__ void k_embed(const int* __restrict__ tok, const float* __restrict__ emb,
                        const float* __restrict__ bnw, const float* __restrict__ bnb,
                        float* __restrict__ out)
{
    const int t = blockIdx.x * blockDim.x + threadIdx.x;
    if (t >= M_ * DMODEL) return;
    const int m = t >> 7, c = t & 127;
    const float s = bnw[c] * rsqrtf(1.0f + 1e-5f);
    out[t] = emb[tok[m] * DMODEL + c] * s + bnb[c];
}

// ---------------- front conv: k=3, pad 1, 128ch -> 128ch, relu ---------------
__global__ void k_conv3(const float* __restrict__ in, const float* __restrict__ wt,
                        const float* __restrict__ bias, float* __restrict__ out)
{
    const int t = blockIdx.x * blockDim.x + threadIdx.x;
    if (t >= M_ * DMODEL) return;
    const int m = t >> 7, o = t & 127;
    const int b = m >> 11, l = m & 2047;
    float acc = bias[o];
#pragma unroll
    for (int k = 0; k < 3; ++k) {
        const int ll = l + k - 1;
        if (ll < 0 || ll >= L_) continue;
        const float4* xr = (const float4*)(in + ((size_t)(b * L_ + ll) << 7));
        const float4* wr = (const float4*)(wt + (k * 128 + o) * 128);
#pragma unroll 8
        for (int i = 0; i < 32; ++i) {
            const float4 x4 = xr[i], w4 = wr[i];
            acc += x4.x * w4.x + x4.y * w4.y + x4.z * w4.z + x4.w * w4.w;
        }
    }
    out[t] = fmaxf(acc, 0.0f);
}

// ---------------- rmsnorm -> bf16 (one wave per 128-wide row) ----------------
__global__ __launch_bounds__(256)
void k_rmsnorm(const float* __restrict__ x, const float* __restrict__ w,
               __bf16* __restrict__ out)
{
    const int lane = threadIdx.x & 31;
    const int row  = blockIdx.x * 8 + (threadIdx.x >> 5);
    const float4 v = *(const float4*)(x + ((size_t)row << 7) + lane * 4);
    float ss = v.x * v.x + v.y * v.y + v.z * v.z + v.w * v.w;
#pragma unroll
    for (int off = 16; off > 0; off >>= 1) ss += __shfl_xor(ss, off, 32);
    const float sc = rsqrtf(ss * (1.0f / 128.0f) + 1e-5f);
    const int c = lane * 4;
    __bf16* o = out + ((size_t)row << 7) + c;
    o[0] = f2bf(v.x * sc * w[c + 0]);
    o[1] = f2bf(v.y * sc * w[c + 1]);
    o[2] = f2bf(v.z * sc * w[c + 2]);
    o[3] = f2bf(v.w * sc * w[c + 3]);
}

// ---------------- depthwise conv k=4 causal + silu (xi = xz[:, :256]) --------
__global__ void k_dwconv4_silu(const __bf16* __restrict__ xz, const float* __restrict__ w,
                               const float* __restrict__ bias, __bf16* __restrict__ out)
{
    const int t = blockIdx.x * blockDim.x + threadIdx.x;
    if (t >= M_ * DINNER) return;
    const int m = t >> 8, d = t & 255;
    const int b = m >> 11, l = m & 2047;
    float acc = bias[d];
#pragma unroll
    for (int k = 0; k < 4; ++k) {
        const int ll = l - 3 + k;
        if (ll >= 0) acc += w[d * 4 + k] * bf2f(xz[((size_t)(b * L_ + ll) << 9) + d]);
    }
    const float s = acc / (1.0f + __expf(-acc));   // silu
    out[t] = f2bf(s);
}

// ---------------- dt = softplus(xdbl[:, :8] @ dtw^T + dtb) -------------------
__global__ void k_dt(const float* __restrict__ xdbl, const float* __restrict__ dw,
                     const float* __restrict__ db, float* __restrict__ out)
{
    const int t = blockIdx.x * blockDim.x + threadIdx.x;
    if (t >= M_ * DINNER) return;
    const int m = t >> 8, d = t & 255;
    float acc = db[d];
    const float* xr = xdbl + (size_t)m * XPN;
#pragma unroll
    for (int r = 0; r < 8; ++r) acc += xr[r] * dw[d * 8 + r];
    out[t] = (acc > 20.0f) ? acc : log1pf(__expf(acc));
}

// ---------------------------------------------------------------------------
// Chunked selective scan. h' = exp(dt*A)*h + (dt*u)*B is affine in h, and the
// per-chunk decay is prod_t exp(dt_t*A) = exp(A * sum_t dt_t).
// Pass 1: per (b,chunk) block, chunk-local recurrence from h=0 -> Q, sum(dt).
// Pass 2: tiny sequential prefix over the 16 chunks -> h_in per chunk.
// Pass 3: re-run chunk recurrence from h_in; y, +u*D, *silu(z) -> bf16 yg.
// ---------------------------------------------------------------------------
__global__ __launch_bounds__(256)
void k_scan1(const float* __restrict__ dt, const __bf16* __restrict__ u,
             const float* __restrict__ xdbl, const float* __restrict__ Aneg,
             float* __restrict__ Q, float* __restrict__ DTS)
{
    __shared__ float sB[DSTATE];
    const int b = blockIdx.x >> 4;
    const int c = blockIdx.x & 15;
    const int d = threadIdx.x;
    float h[DSTATE], Ad[DSTATE];
#pragma unroll
    for (int s = 0; s < DSTATE; ++s) { h[s] = 0.0f; Ad[s] = Aneg[d * DSTATE + s]; }
    float dtsum = 0.0f;

    const int l0 = c * CH;
    for (int i = 0; i < CH; ++i) {
        const size_t row = (size_t)b * L_ + l0 + i;
        if (threadIdx.x < DSTATE) sB[threadIdx.x] = xdbl[row * XPN + 8 + threadIdx.x];
        __syncthreads();
        const float dtv = dt[row * DINNER + d];
        const float du  = dtv * bf2f(u[row * DINNER + d]);
        dtsum += dtv;
#pragma unroll
        for (int s = 0; s < DSTATE; ++s)
            h[s] = __expf(dtv * Ad[s]) * h[s] + du * sB[s];
        __syncthreads();
    }
    const size_t base = ((size_t)(b * NC + c) * DINNER + d);
#pragma unroll
    for (int s = 0; s < DSTATE; ++s) Q[base * DSTATE + s] = h[s];
    DTS[base] = dtsum;
}

__global__ void k_scan2(const float* __restrict__ Q, const float* __restrict__ DTS,
                        const float* __restrict__ Aneg, float* __restrict__ HIN)
{
    const int t = blockIdx.x * blockDim.x + threadIdx.x;
    if (t >= B_ * DINNER) return;
    const int b = t >> 8, d = t & 255;
    float h[DSTATE], Ad[DSTATE];
#pragma unroll
    for (int s = 0; s < DSTATE; ++s) { h[s] = 0.0f; Ad[s] = Aneg[d * DSTATE + s]; }
    for (int c = 0; c < NC; ++c) {
        const size_t base = ((size_t)(b * NC + c) * DINNER + d);
        const float dts = DTS[base];
#pragma unroll
        for (int s = 0; s < DSTATE; ++s) {
            HIN[base * DSTATE + s] = h[s];
            h[s] = __expf(Ad[s] * dts) * h[s] + Q[base * DSTATE + s];
        }
    }
}

__global__ __launch_bounds__(256)
void k_scan3(const float* __restrict__ dt, const __bf16* __restrict__ u,
             const float* __restrict__ xdbl, const __bf16* __restrict__ xz,
             const float* __restrict__ Aneg, const float* __restrict__ Dv,
             const float* __restrict__ HIN, __bf16* __restrict__ yg)
{
    __shared__ float sBC[2 * DSTATE];          // [0..15]=B_t, [16..31]=C_t
    const int b = blockIdx.x >> 4;
    const int c = blockIdx.x & 15;
    const int d = threadIdx.x;
    const size_t base = ((size_t)(b * NC + c) * DINNER + d);
    float h[DSTATE], Ad[DSTATE];
#pragma unroll
    for (int s = 0; s < DSTATE; ++s) {
        h[s]  = HIN[base * DSTATE + s];
        Ad[s] = Aneg[d * DSTATE + s];
    }
    const float dd = Dv[d];

    const int l0 = c * CH;
    for (int i = 0; i < CH; ++i) {
        const size_t row = (size_t)b * L_ + l0 + i;
        if (threadIdx.x < 2 * DSTATE) sBC[threadIdx.x] = xdbl[row * XPN + 8 + threadIdx.x];
        __syncthreads();
        const float dtv = dt[row * DINNER + d];
        const float uv  = bf2f(u[row * DINNER + d]);
        const float du  = dtv * uv;
        float y = 0.0f;
#pragma unroll
        for (int s = 0; s < DSTATE; ++s) {
            h[s] = __expf(dtv * Ad[s]) * h[s] + du * sBC[s];
            y += h[s] * sBC[DSTATE + s];
        }
        y += uv * dd;
        const float z = bf2f(xz[(row << 9) + DINNER + d]);
        const float g = z / (1.0f + __expf(-z));
        yg[row * DINNER + d] = f2bf(y * g);
        __syncthreads();
    }
}

// ---------------- masked mean pool ------------------------------------------
__global__ void k_pool(const float* __restrict__ x, const float* __restrict__ mask,
                       float* __restrict__ out)
{
    const int t = blockIdx.x * blockDim.x + threadIdx.x;
    if (t >= B_ * DMODEL) return;
    const int b = t >> 7, c = t & 127;
    float s = 0.0f, ms = 0.0f;
    for (int l = 0; l < L_; ++l) {
        const float mk = mask[b * L_ + l];
        s  += x[(((size_t)b * L_ + l) << 7) + c] * mk;
        ms += mk;
    }
    out[t] = s / fmaxf(ms, 1e-9f);
}

// ---------------- tiny FC head ----------------------------------------------
__global__ void k_fc(const float* __restrict__ in, const float* __restrict__ w,
                     const float* __restrict__ bias, float* __restrict__ out,
                     int Mm, int Nn, int Kk, int doRelu)
{
    const int t = blockIdx.x * blockDim.x + threadIdx.x;
    if (t >= Mm * Nn) return;
    const int m = t / Nn, o = t % Nn;
    float acc = bias[o];
    for (int k = 0; k < Kk; ++k) acc += in[m * Kk + k] * w[o * Kk + k];
    out[t] = doRelu ? fmaxf(acc, 0.0f) : acc;
}

// ---------------------------------------------------------------------------
extern "C" void kernel_launch(void* const* d_in, const int* in_sizes, int n_in,
                              void* d_out, int out_size, void* d_ws, size_t ws_size,
                              hipStream_t stream)
{
    (void)in_sizes; (void)n_in; (void)out_size; (void)ws_size;
    const int*   tokens = (const int*)  d_in[0];
    const float* mask   = (const float*)d_in[1];
    const float* emb    = (const float*)d_in[2];
    const float* bn_w   = (const float*)d_in[3];
    const float* bn_b   = (const float*)d_in[4];
    const float* conv_w = (const float*)d_in[5];
    const float* conv_b = (const float*)d_in[6];
    const float* ip_w   = (const float*)d_in[7];
    const float* c1w    = (const float*)d_in[8];
    const float* c1b    = (const float*)d_in[9];
    const float* xp_w   = (const float*)d_in[10];
    const float* dtp_w  = (const float*)d_in[11];
    const float* dtp_b  = (const float*)d_in[12];
    const float* A_log  = (const float*)d_in[13];
    const float* Dp     = (const float*)d_in[14];
    const float* op_w   = (const float*)d_in[15];
    const float* norm_w = (const float*)d_in[16];
    const float* l1w    = (const float*)d_in[17];
    const float* l1b    = (const float*)d_in[18];
    const float* l2w    = (const float*)d_in[19];
    const float* l2b    = (const float*)d_in[20];
    const float* l3w    = (const float*)d_in[21];
    const float* l3b    = (const float*)d_in[22];

    // workspace carve-out (~565 MB total)
    char* ws = (char*)d_ws;
    size_t off = 0;
    auto carve = [&](size_t bytes) -> void* {
        void* p = ws + off;
        off = (off + bytes + 255) & ~(size_t)255;
        return p;
    };
    float*  pX    = (float*) carve((size_t)M_ * 128 * 4);   // residual stream
    __bf16* pXN   = (__bf16*)carve((size_t)M_ * 128 * 2);   // normed bf16
    __bf16* pXZ   = (__bf16*)carve((size_t)M_ * 512 * 2);   // in_proj out (xi|z)
    __bf16* pXC   = (__bf16*)carve((size_t)M_ * 256 * 2);   // conv+silu out
    float*  pXDBL = (float*) carve((size_t)M_ * XPN * 4);   // x_proj out (padded)
    float*  pDT   = (float*) carve((size_t)M_ * 256 * 4);   // softplus(dt)
    __bf16* pYG   = (__bf16*)carve((size_t)M_ * 256 * 2);   // gated scan out
    float*  pQ    = (float*) carve((size_t)B_ * NC * DINNER * DSTATE * 4);
    float*  pHIN  = (float*) carve((size_t)B_ * NC * DINNER * DSTATE * 4);
    float*  pDTS  = (float*) carve((size_t)B_ * NC * DINNER * 4);
    float*  pPOOL = (float*) carve((size_t)B_ * 128 * 4);
    float*  pH1   = (float*) carve((size_t)B_ * 512 * 4);
    float*  pH2   = (float*) carve((size_t)B_ * 512 * 4);
    __bf16* wIP   = (__bf16*)carve((size_t)NL * 512 * 128 * 2);
    __bf16* wXP   = (__bf16*)carve((size_t)NL * XPN * 256 * 2);
    __bf16* wOP   = (__bf16*)carve((size_t)NL * 128 * 256 * 2);
    float*  pA    = (float*) carve((size_t)NL * 256 * 16 * 4);
    float*  pCWT  = (float*) carve((size_t)3 * 128 * 128 * 4);
    float*  pTMP  = pDT;   // embed temp aliases pDT (disjoint lifetime)

    const int TPB = 256;
    const int prepN = NL*512*128 + NL*XPN*256 + NL*128*256 + NL*256*16 + 3*128*128;

    k_prep<<<(prepN + TPB - 1) / TPB, TPB, 0, stream>>>(
        ip_w, xp_w, op_w, A_log, conv_w, wIP, wXP, wOP, pA, pCWT);
    k_embed<<<(M_ * 128) / TPB, TPB, 0, stream>>>(tokens, emb, bn_w, bn_b, pTMP);
    k_conv3<<<(M_ * 128) / TPB, TPB, 0, stream>>>(pTMP, pCWT, conv_b, pX);

    for (int i = 0; i < NL; ++i) {
        const float* Ai = pA + i * 256 * 16;
        k_rmsnorm<<<M_ / 8, TPB, 0, stream>>>(pX, norm_w + i * 128, pXN);
        // in_proj: [M,128] x [512,128]^T -> bf16 [M,512]   (16x64 per wave)
        gemm_wmma_bf16<128, 4, true, false><<<(M_/16)*(512/64)/8, TPB, 0, stream>>>(
            pXN, wIP + (size_t)i * 512 * 128, pXZ, nullptr, M_, 512);
        k_dwconv4_silu<<<(M_ * 256) / TPB, TPB, 0, stream>>>(
            pXZ, c1w + i * 256 * 4, c1b + i * 256, pXC);
        // x_proj: [M,256] x [48,256]^T -> f32 [M,48]        (16x48 per wave)
        gemm_wmma_bf16<256, 3, false, false><<<(M_/16)/8, TPB, 0, stream>>>(
            pXC, wXP + (size_t)i * XPN * 256, pXDBL, nullptr, M_, XPN);
        k_dt<<<(M_ * 256) / TPB, TPB, 0, stream>>>(
            pXDBL, dtp_w + i * 256 * 8, dtp_b + i * 256, pDT);
        // chunked scan: 1024-block passes around a tiny sequential prefix
        k_scan1<<<B_ * NC, 256, 0, stream>>>(pDT, pXC, pXDBL, Ai, pQ, pDTS);
        k_scan2<<<(B_ * DINNER) / TPB, TPB, 0, stream>>>(pQ, pDTS, Ai, pHIN);
        k_scan3<<<B_ * NC, 256, 0, stream>>>(
            pDT, pXC, pXDBL, pXZ, Ai, Dp + i * 256, pHIN, pYG);
        // out_proj: [M,256] x [128,256]^T + residual -> f32 pX (in place)
        gemm_wmma_bf16<256, 4, false, true><<<(M_/16)*(128/64)/8, TPB, 0, stream>>>(
            pYG, wOP + (size_t)i * 128 * 256, pX, pX, M_, 128);
    }

    k_pool<<<(B_ * 128) / TPB, TPB, 0, stream>>>(pX, mask, pPOOL);
    k_fc<<<(B_ * 512) / TPB, TPB, 0, stream>>>(pPOOL, l1w, l1b, pH1, B_, 512, 128, 1);
    k_fc<<<(B_ * 512) / TPB, TPB, 0, stream>>>(pH1,   l2w, l2b, pH2, B_, 512, 512, 1);
    k_fc<<<1, TPB, 0, stream>>>(pH2, l3w, l3b, (float*)d_out, B_, 3, 512, 0);
}
